// CrossAttention_257698038371
// MI455X (gfx1250) — matrix operand order repaired and checked by reading, not compiled, not run on hardware
//
#include <hip/hip_runtime.h>
#include <hip/hip_bf16.h>

typedef __attribute__((ext_vector_type(16))) __bf16 v16bf;
typedef __attribute__((ext_vector_type(8)))  __bf16 v8bf;
typedef __attribute__((ext_vector_type(8)))  float  v8f;
typedef __attribute__((ext_vector_type(4)))  float  v4f;
typedef int v4i_gcc __attribute__((vector_size(16)));   // matches builtin param type

#define B_    4
#define N_    1024
#define C_    1024
#define H_    16
#define HD_   128
#define S2_   2048              // 2N
#define ROWS_ 8192              // B*2N
#define D3_   3072              // 3C
#define OUT_  1024
#define SCALE_ 0.125f           // (C/H)^-0.5 = 64^-0.5

#ifndef __has_builtin
#define __has_builtin(x) 0
#endif

#if __has_builtin(__builtin_amdgcn_global_load_async_to_lds_b128)
#define HAVE_ASYNC_LDS 1
#endif

// 16-byte global -> LDS copy: async DMA path on CDNA5 (ASYNCcnt-tracked),
// plain register copy fallback otherwise.
__device__ __forceinline__ void stage16(const void* g, void* l) {
#ifdef HAVE_ASYNC_LDS
    __builtin_amdgcn_global_load_async_to_lds_b128(
        (__attribute__((address_space(1))) v4i_gcc*)(uintptr_t)g,
        (__attribute__((address_space(3))) v4i_gcc*)(unsigned)(uintptr_t)l,
        0, 0);
#else
    *(v8bf*)l = *(const v8bf*)g;
#endif
}

__device__ __forceinline__ void stage_wait() {
#ifdef HAVE_ASYNC_LDS
#if __has_builtin(__builtin_amdgcn_s_wait_asynccnt)
    __builtin_amdgcn_s_wait_asynccnt(0);
#else
    asm volatile("s_wait_asynccnt 0x0" ::: "memory");
#endif
#endif
}

// ---------------- WMMA fragment loaders (bf16, 16x16x32) ----------------
// A-matrix 16x32 layout: lane L (m=L&15) holds K = {g*8..g*8+7, g*8+16..g*8+23}, g=L>>4
__device__ __forceinline__ v16bf frag_a(const __bf16* base, int row, int k0, int lane, int stride) {
    const __bf16* p = base + (size_t)(row + (lane & 15)) * stride + k0 + ((lane >> 4) << 3);
    union { v16bf v; v8bf h[2]; } u;
    u.h[0] = *(const v8bf*)(p);
    u.h[1] = *(const v8bf*)(p + 16);
    return u.v;
}
// B-matrix 32x16 layout: lane L (n=L&15) holds K = {g*8..g*8+7, g*8+16..g*8+23}, g=L>>4
__device__ __forceinline__ v16bf frag_b(const __bf16* base, int row, int k0, int lane, int stride) {
    const __bf16* p = base + (size_t)(row + (lane & 15)) * stride + k0 + ((lane >> 4) << 3);
    union { v16bf v; v8bf h[2]; } u;
    u.h[0] = *(const v8bf*)(p);
    u.h[1] = *(const v8bf*)(p + 16);
    return u.v;
}

#define WMMA_BF16(a, b, c) \
    __builtin_amdgcn_wmma_f32_16x16x32_bf16(false, (a), false, (b), (short)0, (c), false, false)

// ============================================================================
// Kernel 1: QKV = concat(x, context) @ w_qkv^T      [8192,1024]x[3072,1024]^T
// Output written as bf16 into workspace.
// ============================================================================
__global__ __launch_bounds__(256)
void qkv_gemm(const float* __restrict__ x, const float* __restrict__ ctx,
              const float* __restrict__ w, __bf16* __restrict__ qkv)
{
    __shared__ __bf16 As[128][40];
    __shared__ __bf16 Bs[128][40];
    const int tid  = threadIdx.x;
    const int lane = tid & 31, wave = tid >> 5;
    const int wm = wave & 1, wn = wave >> 1;          // 2x4 wave grid
    const int rowBase = blockIdx.y * 128;
    const int colBase = blockIdx.x * 128;

    v8f zero = {};
    v8f acc[4][2];
    #pragma unroll
    for (int i = 0; i < 4; i++)
        #pragma unroll
        for (int j = 0; j < 2; j++) acc[i][j] = zero;

    for (int k0 = 0; k0 < C_; k0 += 32) {
        // stage A: 128x32 f32 -> bf16 (xc rows, concat of x / context)
        #pragma unroll
        for (int i = 0; i < 4; i++) {
            int slot = tid + i * 256;                 // 1024 float4 slots
            int r  = slot >> 3;
            int c4 = (slot & 7) << 2;
            int gr = rowBase + r;
            int b = gr >> 11, s = gr & 2047;
            const float* src = (s < N_) ? (x   + (size_t)(b * N_ + s)        * C_)
                                        : (ctx + (size_t)(b * N_ + (s - N_)) * C_);
            v4f v = *(const v4f*)(src + k0 + c4);
            __builtin_prefetch(src + k0 + c4 + 32, 0, 1);   // next K slab
            As[r][c4 + 0] = (__bf16)v.x;
            As[r][c4 + 1] = (__bf16)v.y;
            As[r][c4 + 2] = (__bf16)v.z;
            As[r][c4 + 3] = (__bf16)v.w;
        }
        // stage B: 128x32 f32 -> bf16 (w_qkv rows)
        #pragma unroll
        for (int i = 0; i < 4; i++) {
            int slot = tid + i * 256;
            int r  = slot >> 3;
            int c4 = (slot & 7) << 2;
            const float* src = w + (size_t)(colBase + r) * C_ + k0 + c4;
            v4f v = *(const v4f*)src;
            __builtin_prefetch(src + 32, 0, 1);
            Bs[r][c4 + 0] = (__bf16)v.x;
            Bs[r][c4 + 1] = (__bf16)v.y;
            Bs[r][c4 + 2] = (__bf16)v.z;
            Bs[r][c4 + 3] = (__bf16)v.w;
        }
        __syncthreads();
        v16bf bfr[2];
        #pragma unroll
        for (int j = 0; j < 2; j++) bfr[j] = frag_b(&Bs[0][0], wn * 32 + j * 16, 0, lane, 40);
        #pragma unroll
        for (int i = 0; i < 4; i++) {
            v16bf afr = frag_a(&As[0][0], wm * 64 + i * 16, 0, lane, 40);
            #pragma unroll
            for (int j = 0; j < 2; j++) acc[i][j] = WMMA_BF16(afr, bfr[j], acc[i][j]);
        }
        __syncthreads();
    }

    const int mlo = (lane >> 4) * 8;
    const int nn  = lane & 15;
    #pragma unroll
    for (int i = 0; i < 4; i++)
        #pragma unroll
        for (int j = 0; j < 2; j++) {
            int col = colBase + wn * 32 + j * 16 + nn;
            #pragma unroll
            for (int r = 0; r < 8; r++) {
                int row = rowBase + wm * 64 + i * 16 + r + mlo;
                qkv[(size_t)row * D3_ + col] = (__bf16)acc[i][j][r];
            }
        }
}

// ============================================================================
// Kernel 2a: S = SCALE * Q @ K^T per (b,h), written f32 into attn region.
// Q[b,h,n,e] = QKV[b*2N + 2n, h*128+e]
// K[b,h,n,e] = h<8 ? QKV[b*2N+2n, 2048+h*128+e] : QKV[b*2N+2n+1, h*128+e-1024]
// Q/K staging is a straight bf16 copy -> async global->LDS DMA.
// ============================================================================
__global__ __launch_bounds__(256)
void attn_scores(const __bf16* __restrict__ qkv, float* __restrict__ attn)
{
    __shared__ __bf16 Qs[128][136];   // full 128 x 128 (HD) Q tile
    __shared__ __bf16 Ks[128][40];    // 128 x 32 e-chunk of K tile
    const int tid  = threadIdx.x;
    const int lane = tid & 31, wave = tid >> 5;
    const int wm = wave & 1, wn = wave >> 1;
    const int bh = blockIdx.z;
    const int b = bh >> 4, h = bh & 15;
    const int qBase = blockIdx.y * 128;
    const int kBase = blockIdx.x * 128;

    // stage Q tile once: 128 rows x 16 chunks of 8 bf16 (async DMA)
    #pragma unroll
    for (int i = 0; i < 8; i++) {
        int slot = tid + i * 256;                  // 2048 slots
        int r  = slot >> 4;
        int c8 = (slot & 15) << 3;
        const __bf16* src = qkv + (size_t)(b * S2_ + 2 * (qBase + r)) * D3_ + h * HD_ + c8;
        stage16(src, &Qs[r][c8]);
    }

    v8f zero = {};
    v8f acc[4][2];
    #pragma unroll
    for (int i = 0; i < 4; i++)
        #pragma unroll
        for (int j = 0; j < 2; j++) acc[i][j] = zero;

    for (int e0 = 0; e0 < HD_; e0 += 32) {
        // stage K e-chunk: 128 rows x 32 bf16 (async DMA)
        #pragma unroll
        for (int i = 0; i < 2; i++) {
            int slot = tid + i * 256;              // 512 slots of 8 bf16
            int r  = slot >> 2;
            int c8 = (slot & 3) << 3;
            const __bf16* src;
            if (h < 8)
                src = qkv + (size_t)(b * S2_ + 2 * (kBase + r)) * D3_ + 2048 + h * HD_ + e0 + c8;
            else
                src = qkv + (size_t)(b * S2_ + 2 * (kBase + r) + 1) * D3_ + (h * HD_ - 1024) + e0 + c8;
            stage16(src, &Ks[r][c8]);
        }
        stage_wait();
        __syncthreads();
        v16bf bfr[2];
        #pragma unroll
        for (int j = 0; j < 2; j++) bfr[j] = frag_b(&Ks[0][0], wn * 32 + j * 16, 0, lane, 40);
        #pragma unroll
        for (int i = 0; i < 4; i++) {
            v16bf afr = frag_a(&Qs[0][0], wm * 64 + i * 16, e0, lane, 136);
            #pragma unroll
            for (int j = 0; j < 2; j++) acc[i][j] = WMMA_BF16(afr, bfr[j], acc[i][j]);
        }
        __syncthreads();
    }

    const int mlo = (lane >> 4) * 8;
    const int nn  = lane & 15;
    float* arow = attn + (size_t)bh * N_ * N_;
    #pragma unroll
    for (int i = 0; i < 4; i++)
        #pragma unroll
        for (int j = 0; j < 2; j++) {
            int n = kBase + wn * 32 + j * 16 + nn;
            #pragma unroll
            for (int r = 0; r < 8; r++) {
                int m = qBase + wm * 64 + i * 16 + r + mlo;
                arow[(size_t)m * N_ + n] = SCALE_ * acc[i][j][r];
            }
        }
}

// ============================================================================
// Kernel 2b: in-place row softmax over last dim (1024), one wave per row.
// ============================================================================
__global__ __launch_bounds__(256)
void softmax_rows(float* __restrict__ attn)
{
    const int lane = threadIdx.x & 31, wave = threadIdx.x >> 5;
    const size_t row = (size_t)blockIdx.x * 8 + wave;
    float* p = attn + row * 1024;

    v4f v[8];
    float m = -3.4e38f;
    #pragma unroll
    for (int i = 0; i < 8; i++) {
        v[i] = *(const v4f*)(p + lane * 4 + i * 128);
        m = fmaxf(m, fmaxf(fmaxf(v[i].x, v[i].y), fmaxf(v[i].z, v[i].w)));
    }
    #pragma unroll
    for (int off = 16; off; off >>= 1) m = fmaxf(m, __shfl_xor(m, off, 32));

    float s = 0.f;
    #pragma unroll
    for (int i = 0; i < 8; i++) {
        v[i].x = __expf(v[i].x - m);
        v[i].y = __expf(v[i].y - m);
        v[i].z = __expf(v[i].z - m);
        v[i].w = __expf(v[i].w - m);
        s += v[i].x + v[i].y + v[i].z + v[i].w;
    }
    #pragma unroll
    for (int off = 16; off; off >>= 1) s += __shfl_xor(s, off, 32);
    float inv = 1.0f / s;

    #pragma unroll
    for (int i = 0; i < 8; i++) {
        v[i] = v[i] * inv;
        *(v4f*)(p + lane * 4 + i * 128) = v[i];
    }
}

// ============================================================================
// Kernel 2c: O = P @ V per (b,h); result stored directly in the transposed
// o2 layout [B, 2N, C] (bf16) for the projection GEMM.
// V[b,h,k,e] = QKV[b*2N + 2k + 1, 1024 + h*128 + e]
// o2[b, 2n + (h>=8), (h%8)*128 + e] = O[b,h,n,e]
// ============================================================================
__global__ __launch_bounds__(256)
void attn_v(const float* __restrict__ attn, const __bf16* __restrict__ qkv,
            __bf16* __restrict__ o2)
{
    __shared__ __bf16 Ps[128][40];    // P tile (A), bf16
    __shared__ __bf16 Vt[128][40];    // V tile transposed: Vt[e][k]
    const int tid  = threadIdx.x;
    const int lane = tid & 31, wave = tid >> 5;
    const int wm = wave & 1, wn = wave >> 1;
    const int bh = blockIdx.y;
    const int b = bh >> 4, h = bh & 15;
    const int mBase = blockIdx.x * 128;
    const float* P = attn + (size_t)bh * N_ * N_;

    v8f zero = {};
    v8f acc[4][2];
    #pragma unroll
    for (int i = 0; i < 4; i++)
        #pragma unroll
        for (int j = 0; j < 2; j++) acc[i][j] = zero;

    for (int k0 = 0; k0 < N_; k0 += 32) {
        // stage P: 128x32 f32 -> bf16
        #pragma unroll
        for (int i = 0; i < 4; i++) {
            int slot = tid + i * 256;
            int r  = slot >> 3;
            int c4 = (slot & 7) << 2;
            v4f v = *(const v4f*)(P + (size_t)(mBase + r) * N_ + k0 + c4);
            Ps[r][c4 + 0] = (__bf16)v.x;
            Ps[r][c4 + 1] = (__bf16)v.y;
            Ps[r][c4 + 2] = (__bf16)v.z;
            Ps[r][c4 + 3] = (__bf16)v.w;
        }
        // stage V transposed: rows k (32) x e (128) -> Vt[e][k]
        #pragma unroll
        for (int i = 0; i < 2; i++) {
            int slot = tid + i * 256;              // 512 slots
            int k  = slot & 31;
            int e0 = (slot >> 5) << 3;
            const __bf16* src = qkv + (size_t)(b * S2_ + 2 * (k0 + k) + 1) * D3_
                                    + 1024 + h * HD_ + e0;
            v8bf vv = *(const v8bf*)src;
            #pragma unroll
            for (int j = 0; j < 8; j++) Vt[e0 + j][k] = vv[j];
        }
        __syncthreads();
        v16bf bfr[2];
        #pragma unroll
        for (int j = 0; j < 2; j++) bfr[j] = frag_b(&Vt[0][0], wn * 32 + j * 16, 0, lane, 40);
        #pragma unroll
        for (int i = 0; i < 4; i++) {
            v16bf afr = frag_a(&Ps[0][0], wm * 64 + i * 16, 0, lane, 40);
            #pragma unroll
            for (int j = 0; j < 2; j++) acc[i][j] = WMMA_BF16(afr, bfr[j], acc[i][j]);
        }
        __syncthreads();
    }

    const int mlo = (lane >> 4) * 8;
    const int nn  = lane & 15;
    #pragma unroll
    for (int i = 0; i < 4; i++)
        #pragma unroll
        for (int j = 0; j < 2; j++) {
            int e   = wn * 32 + j * 16 + nn;
            int col = (h & 7) * HD_ + e;
            #pragma unroll
            for (int r = 0; r < 8; r++) {
                int m   = mBase + wm * 64 + i * 16 + r + mlo;
                int row = b * S2_ + 2 * m + (h >> 3);
                o2[(size_t)row * C_ + col] = (__bf16)acc[i][j][r];
            }
        }
}

// ============================================================================
// Kernel 3: out = o2 @ w_proj^T + b_proj        [8192,1024]x[1024,1024]^T
// A-tile staging is a straight bf16 copy -> async global->LDS DMA.
// ============================================================================
__global__ __launch_bounds__(256)
void proj_gemm(const __bf16* __restrict__ o2, const float* __restrict__ w,
               const float* __restrict__ bias, float* __restrict__ out)
{
    __shared__ __bf16 As[128][40];
    __shared__ __bf16 Bs[128][40];
    const int tid  = threadIdx.x;
    const int lane = tid & 31, wave = tid >> 5;
    const int wm = wave & 1, wn = wave >> 1;
    const int rowBase = blockIdx.y * 128;
    const int colBase = blockIdx.x * 128;

    v8f zero = {};
    v8f acc[4][2];
    #pragma unroll
    for (int i = 0; i < 4; i++)
        #pragma unroll
        for (int j = 0; j < 2; j++) acc[i][j] = zero;

    for (int k0 = 0; k0 < C_; k0 += 32) {
        // stage A: bf16 copy 128x32 (async DMA)
        #pragma unroll
        for (int i = 0; i < 2; i++) {
            int slot = tid + i * 256;              // 512 slots of 8 bf16
            int r  = slot >> 2;
            int c8 = (slot & 3) << 3;
            stage16(o2 + (size_t)(rowBase + r) * C_ + k0 + c8, &As[r][c8]);
        }
        // stage B: w_proj rows f32 -> bf16
        #pragma unroll
        for (int i = 0; i < 4; i++) {
            int slot = tid + i * 256;
            int r  = slot >> 3;
            int c4 = (slot & 7) << 2;
            const float* src = w + (size_t)(colBase + r) * C_ + k0 + c4;
            v4f v = *(const v4f*)src;
            __builtin_prefetch(src + 32, 0, 1);
            Bs[r][c4 + 0] = (__bf16)v.x;
            Bs[r][c4 + 1] = (__bf16)v.y;
            Bs[r][c4 + 2] = (__bf16)v.z;
            Bs[r][c4 + 3] = (__bf16)v.w;
        }
        stage_wait();
        __syncthreads();
        v16bf bfr[2];
        #pragma unroll
        for (int j = 0; j < 2; j++) bfr[j] = frag_b(&Bs[0][0], wn * 32 + j * 16, 0, lane, 40);
        #pragma unroll
        for (int i = 0; i < 4; i++) {
            v16bf afr = frag_a(&As[0][0], wm * 64 + i * 16, 0, lane, 40);
            #pragma unroll
            for (int j = 0; j < 2; j++) acc[i][j] = WMMA_BF16(afr, bfr[j], acc[i][j]);
        }
        __syncthreads();
    }

    const int mlo = (lane >> 4) * 8;
    const int nn  = lane & 15;
    #pragma unroll
    for (int i = 0; i < 4; i++)
        #pragma unroll
        for (int j = 0; j < 2; j++) {
            int col = colBase + wn * 32 + j * 16 + nn;
            float bv = bias[col];
            #pragma unroll
            for (int r = 0; r < 8; r++) {
                int row = rowBase + wm * 64 + i * 16 + r + mlo;
                out[(size_t)row * OUT_ + col] = acc[i][j][r] + bv;
            }
        }
}

// ============================================================================
extern "C" void kernel_launch(void* const* d_in, const int* in_sizes, int n_in,
                              void* d_out, int out_size, void* d_ws, size_t ws_size,
                              hipStream_t stream)
{
    const float* x      = (const float*)d_in[0];
    const float* ctx    = (const float*)d_in[1];
    const float* w_qkv  = (const float*)d_in[2];
    const float* w_proj = (const float*)d_in[3];
    const float* b_proj = (const float*)d_in[4];

    float* out  = (float*)d_out;                          // [B, 2N, OUT]
    float* attn = out + (size_t)ROWS_ * OUT_;             // [B, H, N, N]

    __bf16* qkv = (__bf16*)d_ws;                                          // 48 MB
    __bf16* o2  = (__bf16*)((char*)d_ws + (size_t)ROWS_ * D3_ * sizeof(__bf16)); // +16 MB

    qkv_gemm   <<<dim3(D3_ / 128, ROWS_ / 128), 256, 0, stream>>>(x, ctx, w_qkv, qkv);
    attn_scores<<<dim3(N_ / 128, N_ / 128, B_ * H_), 256, 0, stream>>>(qkv, attn);
    softmax_rows<<<dim3((B_ * H_ * N_) / 8), 256, 0, stream>>>(attn);
    attn_v     <<<dim3(N_ / 128, B_ * H_), 256, 0, stream>>>(attn, qkv, o2);
    proj_gemm  <<<dim3(OUT_ / 128, ROWS_ / 128), 256, 0, stream>>>(o2, w_proj, b_proj, out);
}
